// InferDepthNet_51599737094658
// MI455X (gfx1250) — matrix operand order repaired and checked by reading, not compiled, not run on hardware
//
#include <hip/hip_runtime.h>
#include <hip/hip_bf16.h>
#include <math.h>

// ---------------------------------------------------------------------------
// Problem dimensions (fixed by the reference)
// ---------------------------------------------------------------------------
#define V_  3
#define C_  32
#define H_  128
#define W_  160
#define D_  32

#define ACT_NONE 0
#define ACT_RELU 1
#define ACT_SIG  2
#define ACT_TANH 3

#define KMAX 1152   // max Cin*9 (gru4: 128*9)

typedef __attribute__((ext_vector_type(16))) _Float16 v16h;
typedef __attribute__((ext_vector_type(8)))  float    v8f;
typedef __attribute__((ext_vector_type(4)))  unsigned int u32x4;
typedef __attribute__((ext_vector_type(8)))  int      i32x8;
typedef __attribute__((ext_vector_type(4)))  int      i32x4;

union V16H { v16h v; _Float16 f[16]; };
union V8F  { v8f  v; float    f[8];  };

// ---------------------------------------------------------------------------
// Generic 3x3 conv as implicit GEMM on v_wmma_f32_16x16x32_f16.
//   input = concat(in0 [C0 ch f16, optionally nearest-2x-upsampled], in1 [C1 ch f16])
//   weights f16 in [Cout][Cin*9] order (== OIHW flattened); the block's 16-row
//   weight slab is DMA'd to LDS ONCE via the Tensor Data Mover, then the K-loop
//   touches only LDS.  Per-wave tile: 32 output pixels x 16 output channels
//   (2 M-fragments share one B-fragment -> 2 WMMA per K chunk).
// ---------------------------------------------------------------------------
__global__ __launch_bounds__(256)
void conv3x3_wmma_kernel(const _Float16* __restrict__ in0, int C0, int up0,
                         const _Float16* __restrict__ in1, int C1,
                         const _Float16* __restrict__ w16,
                         const float* __restrict__ bias,
                         float*    __restrict__ outf,   // optional f32 output
                         _Float16* __restrict__ outh,   // optional f16 output
                         int Cout, int Hin, int Win, int stride, int act)
{
    const int Hout = Hin / stride, Wout = Win / stride;
    const int Npix = Hout * Wout;
    const int Cin  = C0 + C1;
    const int Ktot = Cin * 9;
    const int pad  = (stride == 1) ? 1 : 0;   // JAX SAME: lo-pad 1 (s=1), 0 (s=2)
    const int H0   = Hin / up0, W0 = Win / up0;

    const int lane    = threadIdx.x & 31;
    const int wave    = threadIdx.x >> 5;
    const int pixBase = blockIdx.x * 256 + wave * 32;
    const int coBase  = blockIdx.y * 16;

    __shared__ _Float16 Ws[16 * KMAX];      // weight slab: 16 rows x Ktot (pitch Ktot)
    __shared__ _Float16 As[8][32][32];      // per-wave A tile: 32 pixels x 32 K

    // ---- stage the whole weight slab for this Cout tile ONCE ----
#if __has_builtin(__builtin_amdgcn_tensor_load_to_lds) && __has_builtin(__builtin_amdgcn_s_wait_tensorcnt)
    {
        int rows = Cout - coBase; if (rows > 16) rows = 16;
        unsigned long long gaddr =
            (unsigned long long)(uintptr_t)(w16 + (size_t)coBase * Ktot);
        // flat shared addr: low 32 bits are the LDS byte address
        unsigned int lds_addr = (unsigned int)(uintptr_t)&Ws[0];

        u32x4 g0;
        g0[0] = 1u;                                    // count=1, user descriptor
        g0[1] = lds_addr;                              // lds_addr
        g0[2] = (unsigned int)(gaddr & 0xffffffffu);   // global_addr[31:0]
        g0[3] = (unsigned int)((gaddr >> 32) & 0x01ffffffu) | (2u << 30); // addr[56:32] | type=2

        i32x8 g1;
        g1[0] = 0x00010000;                            // data_size=1 (2 bytes)
        g1[1] = (int)(((unsigned)Ktot & 0xffffu) << 16);              // tensor_dim0 lo
        g1[2] = (int)((((unsigned)Ktot >> 16) & 0xffffu)
                      | (((unsigned)Cout & 0xffffu) << 16));          // td0 hi | tensor_dim1 lo
        g1[3] = (int)((((unsigned)Ktot) & 0xffffu) << 16);            // td1 hi=0 | tile_dim0=Ktot
        g1[4] = rows & 0xffff;                                        // tile_dim1 (tile_dim2=0)
        g1[5] = Ktot;                                                 // tensor_dim0_stride lo
        g1[6] = 0;                                                    // stride hi | td1_stride lo
        g1[7] = 0;

        i32x4 g2 = {0, 0, 0, 0};
        i32x4 g3 = {0, 0, 0, 0};
        i32x8 g4 = {0, 0, 0, 0, 0, 0, 0, 0};

        if (wave == 0) {
            __builtin_amdgcn_tensor_load_to_lds(g0, g1, g2, g3, g4, 0);
            __builtin_amdgcn_s_wait_tensorcnt(0);
        }
    }
#else
    // fallback: cooperative staging through VGPRs
    for (int i = threadIdx.x; i < 16 * Ktot; i += 256) {
        int r = i / Ktot, k = i - r * Ktot;
        _Float16 wv = (_Float16)0.0f;
        if (coBase + r < Cout) wv = w16[(size_t)(coBase + r) * Ktot + k];
        Ws[i] = wv;
    }
#endif
    __syncthreads();

    const int half = lane >> 4, mr = lane & 15;
    const bool rowok = (coBase + mr) < Cout;

    V8F acc0 = {};
    V8F acc1 = {};

    for (int k0 = 0; k0 < Ktot; k0 += 32) {
        // ---- stage A tile (each wave stages its own 32x32 im2col patch) ----
        for (int i = lane; i < 1024; i += 32) {
            int m = i >> 5, kk = i & 31;
            int k = k0 + kk;
            _Float16 val = (_Float16)0.0f;
            int pix = pixBase + m;
            if (k < Ktot && pix < Npix) {
                int ci = k / 9, tap = k - ci * 9;
                int ky = tap / 3, kx = tap - ky * 3;
                int yo = pix / Wout, xo = pix - yo * Wout;
                int yi = yo * stride + ky - pad;
                int xi = xo * stride + kx - pad;
                if (yi >= 0 && yi < Hin && xi >= 0 && xi < Win) {
                    if (ci < C0) {   // nearest upsample when up0==2
                        val = in0[((size_t)ci * H0 + (yi / up0)) * W0 + (xi / up0)];
                    } else {
                        val = in1[((size_t)(ci - C0) * Hin + yi) * Win + xi];
                    }
                }
            }
            As[wave][m][kk] = val;
        }
        __syncthreads();

        // ---- pack fragments per the CDNA5 wave32 VGPR layouts ----
        V16H a0, a1, b;
#pragma unroll
        for (int v = 0; v < 8; ++v) {
#pragma unroll
            for (int e = 0; e < 2; ++e) {
                // A 16x32 f16: v0..3 -> K = 2v+e (+8 for lanes 16..31)
                //              v4..7 -> K = 16 + 2(v-4)+e (+8 for lanes 16..31)
                int ka = (v < 4 ? (v * 2 + e) : (16 + (v - 4) * 2 + e)) + half * 8;
                a0.f[v * 2 + e] = As[wave][mr][ka];
                a1.f[v * 2 + e] = As[wave][16 + mr][ka];
                // B 32x16 f16: same K order as A
                int kg = k0 + ka;
                b.f[v * 2 + e] = (rowok && kg < Ktot)
                                   ? Ws[mr * Ktot + kg] : (_Float16)0.0f;
            }
        }
        acc0.v = __builtin_amdgcn_wmma_f32_16x16x32_f16(
                     false, a0.v, false, b.v, (short)0, acc0.v, false, false);
        acc1.v = __builtin_amdgcn_wmma_f32_16x16x32_f16(
                     false, a1.v, false, b.v, (short)0, acc1.v, false, false);
        __syncthreads();
    }

    // ---- epilogue: D fragment lane l, vgpr v -> M = v + 8*(l>=16), N = l%16
    const int co = coBase + mr;
    if (co < Cout) {
        float bval = bias ? bias[co] : 0.0f;
#pragma unroll
        for (int v = 0; v < 8; ++v) {
            int m = v + 8 * half;
#pragma unroll
            for (int t = 0; t < 2; ++t) {
                int pix = pixBase + t * 16 + m;
                if (pix < Npix) {
                    float r = (t ? acc1.f[v] : acc0.f[v]) + bval;
                    if      (act == ACT_RELU) r = fmaxf(r, 0.0f);
                    else if (act == ACT_SIG)  r = 1.0f / (1.0f + __expf(-r));
                    else if (act == ACT_TANH) r = tanhf(r);
                    if (outf) outf[(size_t)co * Npix + pix] = r;
                    if (outh) outh[(size_t)co * Npix + pix] = (_Float16)r;
                }
            }
        }
    }
}

// ---------------------------------------------------------------------------
// Projection prep (1 thread): invP0 via Gauss-Jordan, then for v=1,2:
// pm = P_v @ invP0 ; store rot3x3 + trans3 -> Mt[2][12]
// ---------------------------------------------------------------------------
__global__ void proj_prep_kernel(const float* __restrict__ proj, float* __restrict__ Mt)
{
    if (threadIdx.x != 0 || blockIdx.x != 0) return;
    float A[4][8];
    for (int r = 0; r < 4; ++r)
        for (int c = 0; c < 4; ++c) {
            A[r][c]     = proj[r * 4 + c];
            A[r][4 + c] = (r == c) ? 1.0f : 0.0f;
        }
    for (int col = 0; col < 4; ++col) {
        int piv = col;
        for (int r = col + 1; r < 4; ++r)
            if (fabsf(A[r][col]) > fabsf(A[piv][col])) piv = r;
        if (piv != col)
            for (int c = 0; c < 8; ++c) { float t = A[col][c]; A[col][c] = A[piv][c]; A[piv][c] = t; }
        float inv = 1.0f / A[col][col];
        for (int c = 0; c < 8; ++c) A[col][c] *= inv;
        for (int r = 0; r < 4; ++r) if (r != col) {
            float f = A[r][col];
            for (int c = 0; c < 8; ++c) A[r][c] -= f * A[col][c];
        }
    }
    float inv0[16];
    for (int r = 0; r < 4; ++r)
        for (int c = 0; c < 4; ++c) inv0[r * 4 + c] = A[r][4 + c];
    for (int v = 1; v < V_; ++v) {
        const float* P = proj + v * 16;
        float pm[16];
        for (int r = 0; r < 4; ++r)
            for (int c = 0; c < 4; ++c) {
                float s = 0.0f;
                for (int k = 0; k < 4; ++k) s += P[r * 4 + k] * inv0[k * 4 + c];
                pm[r * 4 + c] = s;
            }
        float* o = Mt + (v - 1) * 12;
        for (int r = 0; r < 3; ++r) {
            o[r * 3 + 0] = pm[r * 4 + 0];
            o[r * 3 + 1] = pm[r * 4 + 1];
            o[r * 3 + 2] = pm[r * 4 + 2];
            o[9 + r]     = pm[r * 4 + 3];
        }
    }
}

// ---------------------------------------------------------------------------
// Homography warp + bilinear sample (zero pad) + variance across 3 views.
// Emits the cost volume directly in f16 (WMMA feed precision).
// ---------------------------------------------------------------------------
__global__ __launch_bounds__(256)
void warp_var_kernel(const float* __restrict__ feats,   // [3][32][H][W] f32
                     const float* __restrict__ Mt,      // [2][12]
                     const float* __restrict__ depth_d, // [H][W]
                     _Float16* __restrict__ var16)      // [32][H][W] f16
{
    const int HW = H_ * W_;
    int pix = blockIdx.x * blockDim.x + threadIdx.x;
    if (pix >= HW) return;

    float x = (float)(pix % W_), y = (float)(pix / W_);
    float d = depth_d[pix];

    int   idx[2][4];
    float wgt[2][4];
#pragma unroll
    for (int v = 0; v < 2; ++v) {
        const float* m = Mt + v * 12;
        float rx = m[0] * x + m[1] * y + m[2];
        float ry = m[3] * x + m[4] * y + m[5];
        float rz = m[6] * x + m[7] * y + m[8];
        float pxn = rx * d + m[9], pyn = ry * d + m[10], pzn = rz * d + m[11];
        float px = pxn / pzn, py = pyn / pzn;
        float x0 = floorf(px), y0 = floorf(py);
        float fx = px - x0, fy = py - y0;
        int xi = (int)x0, yi = (int)y0;
#pragma unroll
        for (int j = 0; j < 2; ++j)
#pragma unroll
            for (int i = 0; i < 2; ++i) {
                int xx = xi + i, yy = yi + j;
                bool valid = (xx >= 0) && (xx < W_) && (yy >= 0) && (yy < H_);
                int cx = min(max(xx, 0), W_ - 1);
                int cy = min(max(yy, 0), H_ - 1);
                idx[v][j * 2 + i] = cy * W_ + cx;
                float wv = (i ? fx : 1.0f - fx) * (j ? fy : 1.0f - fy);
                wgt[v][j * 2 + i] = valid ? wv : 0.0f;
            }
    }

    __builtin_prefetch(feats + (size_t)C_ * HW + idx[0][0], 0, 1); // global_prefetch_b8

    for (int c = 0; c < C_; ++c) {
        float s = feats[(size_t)c * HW + pix];   // reference view
        float q = s * s;
#pragma unroll
        for (int v = 0; v < 2; ++v) {
            const float* fv = feats + ((size_t)(v + 1) * C_ + c) * HW;
            float w = wgt[v][0] * fv[idx[v][0]] + wgt[v][1] * fv[idx[v][1]]
                    + wgt[v][2] * fv[idx[v][2]] + wgt[v][3] * fv[idx[v][3]];
            s += w; q += w * w;
        }
        float mean = s * (1.0f / 3.0f);
        var16[(size_t)c * HW + pix] = (_Float16)(q * (1.0f / 3.0f) - mean * mean);
    }
}

// ---------------------------------------------------------------------------
// Elementwise helpers
// ---------------------------------------------------------------------------
__global__ void f32_to_f16_kernel(const float* __restrict__ src,
                                  _Float16* __restrict__ dst, int n)
{
    int i = blockIdx.x * blockDim.x + threadIdx.x;
    if (i < n) dst[i] = (_Float16)src[i];
}

__global__ void zero_kernel(float* __restrict__ p, int n)
{
    int i = blockIdx.x * blockDim.x + threadIdx.x;
    if (i < n) p[i] = 0.0f;
}

__global__ void zero16_kernel(_Float16* __restrict__ p, int n)
{
    int i = blockIdx.x * blockDim.x + threadIdx.x;
    if (i < n) p[i] = (_Float16)0.0f;
}

// rh = r * h  (conv-only consumer -> f16)
__global__ void mul16_kernel(const float* __restrict__ a, const float* __restrict__ b,
                             _Float16* __restrict__ o, int n)
{
    int i = blockIdx.x * blockDim.x + threadIdx.x;
    if (i < n) o[i] = (_Float16)(a[i] * b[i]);
}

// h = (1-z)*h + z*hh ; also refresh the f16 mirror used as conv input
__global__ void gruout_kernel(const float* __restrict__ z, const float* __restrict__ hh,
                              float* __restrict__ h, _Float16* __restrict__ h16, int n)
{
    int i = blockIdx.x * blockDim.x + threadIdx.x;
    if (i < n) {
        float zi = z[i];
        float hn = (1.0f - zi) * h[i] + zi * hh[i];
        h[i]   = hn;
        h16[i] = (_Float16)hn;
    }
}

__global__ void accum_kernel(const float* __restrict__ reg, const float* __restrict__ depth_d,
                             float* __restrict__ exp_sum, float* __restrict__ depth_img,
                             float* __restrict__ max_prob, int n)
{
    int i = blockIdx.x * blockDim.x + threadIdx.x;
    if (i < n) {
        float p = __expf(reg[i]);
        exp_sum[i]   += p;
        depth_img[i] += depth_d[i] * p;
        max_prob[i]   = fmaxf(max_prob[i], p);
    }
}

__global__ void finalize_kernel(const float* __restrict__ depth_img,
                                const float* __restrict__ max_prob,
                                const float* __restrict__ exp_sum,
                                float* __restrict__ out, int n)
{
    int i = blockIdx.x * blockDim.x + threadIdx.x;
    if (i < n) {
        float denom = exp_sum[i] + 1e-10f;
        out[i]     = depth_img[i] / denom;
        out[n + i] = max_prob[i] / denom;
    }
}

// ---------------------------------------------------------------------------
// Host orchestration
// ---------------------------------------------------------------------------
extern "C" void kernel_launch(void* const* d_in, const int* in_sizes, int n_in,
                              void* d_out, int out_size, void* d_ws, size_t ws_size,
                              hipStream_t stream)
{
    (void)in_sizes; (void)out_size; (void)ws_size;
    if (n_in < 42) return;

    const float* feats = (const float*)d_in[0];   // [3][1][32][128][160]
    const float* proj  = (const float*)d_in[1];   // [1][3][4][4]
    const float* depth = (const float*)d_in[2];   // [1][32][128][160]
    // d_in[3] = num_depth scalar (== D_)

    // params flattened with sorted keys (jax pytree order):
    const float* bd1  = (const float*)d_in[4];
    const float* bd2  = (const float*)d_in[6];
    const float* bd3  = (const float*)d_in[8];
    const float* g1bh = (const float*)d_in[10]; const float* g1br = (const float*)d_in[11]; const float* g1bz = (const float*)d_in[12];
    const float* g2bh = (const float*)d_in[16]; const float* g2br = (const float*)d_in[17]; const float* g2bz = (const float*)d_in[18];
    const float* g3bh = (const float*)d_in[22]; const float* g3br = (const float*)d_in[23]; const float* g3bz = (const float*)d_in[24];
    const float* g4bh = (const float*)d_in[28]; const float* g4br = (const float*)d_in[29]; const float* g4bz = (const float*)d_in[30];
    const float* bob  = (const float*)d_in[34];
    const float* bu1  = (const float*)d_in[36];
    const float* bu2  = (const float*)d_in[38];
    const float* bu3  = (const float*)d_in[40];

    const int HW1 = H_ * W_;                               // 20480
    const int H2 = H_ / 2, W2 = W_ / 2, HW2 = H2 * W2;     // 64x80
    const int H3 = H_ / 4, W3 = W_ / 4, HW3 = H3 * W3;     // 32x40
    const int H4 = H_ / 8, W4 = W_ / 8, HW4 = H4 * W4;     // 16x20

    // ---- workspace carve-up ----
    char* ws = (char*)d_ws;
    size_t off = 0;
    auto alloc = [&](size_t bytes) -> char* {
        char* p = ws + off;
        off = (off + bytes + 255) & ~(size_t)255;
        return p;
    };
    // f32 state / elementwise buffers
    float* h1    = (float*)alloc(sizeof(float) * 8  * HW1);
    float* h2    = (float*)alloc(sizeof(float) * 16 * HW2);
    float* h3    = (float*)alloc(sizeof(float) * 32 * HW3);
    float* h4    = (float*)alloc(sizeof(float) * 64 * HW4);
    float* zbuf  = (float*)alloc(sizeof(float) * 8 * HW1);
    float* rbuf  = (float*)alloc(sizeof(float) * 8 * HW1);
    float* hhbuf = (float*)alloc(sizeof(float) * 8 * HW1);
    float* regb  = (float*)alloc(sizeof(float) * HW1);
    float* esum  = (float*)alloc(sizeof(float) * HW1);
    float* dimg  = (float*)alloc(sizeof(float) * HW1);
    float* mprob = (float*)alloc(sizeof(float) * HW1);
    float* Mt    = (float*)alloc(sizeof(float) * 32);
    // f16 conv-input buffers
    _Float16* var16 = (_Float16*)alloc(sizeof(_Float16) * (size_t)C_ * HW1);
    _Float16* h1f   = (_Float16*)alloc(sizeof(_Float16) * 8  * HW1);
    _Float16* h2f   = (_Float16*)alloc(sizeof(_Float16) * 16 * HW2);
    _Float16* h3f   = (_Float16*)alloc(sizeof(_Float16) * 32 * HW3);
    _Float16* h4f   = (_Float16*)alloc(sizeof(_Float16) * 64 * HW4);
    _Float16* rh16  = (_Float16*)alloc(sizeof(_Float16) * 8 * HW1);
    _Float16* x16   = (_Float16*)alloc(sizeof(_Float16) * 16 * HW2); // x2/x3/x4 reuse
    _Float16* u3f   = (_Float16*)alloc(sizeof(_Float16) * 32 * HW3);
    _Float16* u2f   = (_Float16*)alloc(sizeof(_Float16) * 16 * HW2);
    _Float16* u1f   = (_Float16*)alloc(sizeof(_Float16) * 8  * HW1);

    // ---- f16 weight conversion (OIHW flat == [Cout][Cin*9]) ----
    struct WSp { int wi, cout, cin; };
    const WSp wspec[19] = {
        {5,16,8},{7,32,16},{9,64,32},
        {13,8,40},{14,8,40},{15,8,40},
        {19,16,32},{20,16,32},{21,16,32},
        {25,32,64},{26,32,64},{27,32,64},
        {31,64,128},{32,64,128},{33,64,128},
        {35,1,8},
        {37,8,24},{39,16,48},{41,32,96}};
    _Float16* wp[19];
    for (int i = 0; i < 19; ++i) {
        int n = wspec[i].cout * wspec[i].cin * 9;
        wp[i] = (_Float16*)alloc(sizeof(_Float16) * n);
        f32_to_f16_kernel<<<(n + 255) / 256, 256, 0, stream>>>(
            (const float*)d_in[wspec[i].wi], wp[i], n);
    }
    _Float16 *wd1 = wp[0], *wd2 = wp[1], *wd3 = wp[2];
    _Float16 *g1wh = wp[3],  *g1wr = wp[4],  *g1wz = wp[5];
    _Float16 *g2wh = wp[6],  *g2wr = wp[7],  *g2wz = wp[8];
    _Float16 *g3wh = wp[9],  *g3wr = wp[10], *g3wz = wp[11];
    _Float16 *g4wh = wp[12], *g4wr = wp[13], *g4wz = wp[14];
    _Float16 *wow = wp[15], *wu1 = wp[16], *wu2 = wp[17], *wu3 = wp[18];

    // ---- init state (stateless across calls) ----
    auto zero = [&](float* p, int n) {
        zero_kernel<<<(n + 255) / 256, 256, 0, stream>>>(p, n);
    };
    auto zero16 = [&](_Float16* p, int n) {
        zero16_kernel<<<(n + 255) / 256, 256, 0, stream>>>(p, n);
    };
    zero(h1, 8 * HW1); zero(h2, 16 * HW2); zero(h3, 32 * HW3); zero(h4, 64 * HW4);
    zero16(h1f, 8 * HW1); zero16(h2f, 16 * HW2); zero16(h3f, 32 * HW3); zero16(h4f, 64 * HW4);
    zero(esum, HW1); zero(dimg, HW1); zero(mprob, HW1);

    proj_prep_kernel<<<1, 32, 0, stream>>>(proj, Mt);

    auto conv = [&](const _Float16* in0, int C0, int up0, const _Float16* in1, int C1,
                    const _Float16* w, const float* b,
                    float* outf, _Float16* outh, int Cout,
                    int Hin, int Win, int stride, int act) {
        int Npix = (Hin / stride) * (Win / stride);
        dim3 g((Npix + 255) / 256, (Cout + 15) / 16);
        conv3x3_wmma_kernel<<<g, 256, 0, stream>>>(
            in0, C0, up0, in1, C1, w, b, outf, outh, Cout, Hin, Win, stride, act);
    };

    auto gru = [&](const _Float16* x16in, int Cx, float* h, _Float16* h16, int Ch,
                   const _Float16* wz, const float* bz,
                   const _Float16* wr, const float* br,
                   const _Float16* wh, const float* bh,
                   int Hl, int Wl) {
        int n = Ch * Hl * Wl;
        conv(x16in, Cx, 1, h16, Ch, wz, bz, zbuf, nullptr, Ch, Hl, Wl, 1, ACT_SIG);
        conv(x16in, Cx, 1, h16, Ch, wr, br, rbuf, nullptr, Ch, Hl, Wl, 1, ACT_SIG);
        mul16_kernel<<<(n + 255) / 256, 256, 0, stream>>>(rbuf, h, rh16, n);
        conv(x16in, Cx, 1, rh16, Ch, wh, bh, hhbuf, nullptr, Ch, Hl, Wl, 1, ACT_TANH);
        gruout_kernel<<<(n + 255) / 256, 256, 0, stream>>>(zbuf, hhbuf, h, h16, n);
    };

    // ---- 32 sequential depth-plane steps ----
    for (int d = 0; d < D_; ++d) {
        const float* depth_d = depth + (size_t)d * HW1;

        warp_var_kernel<<<(HW1 + 255) / 256, 256, 0, stream>>>(feats, Mt, depth_d, var16);

        // gru1 @ full res: x = var (32ch), h = h1 (8ch)
        gru(var16, 32, h1, h1f, 8, g1wz, g1bz, g1wr, g1br, g1wh, g1bh, H_, W_);
        // down 1: relu(conv s2) 8->16 @ 64x80
        conv(h1f, 8, 1, nullptr, 0, wd1, bd1, nullptr, x16, 16, H_, W_, 2, ACT_RELU);
        gru(x16, 16, h2, h2f, 16, g2wz, g2bz, g2wr, g2br, g2wh, g2bh, H2, W2);
        // down 2: 16->32 @ 32x40
        conv(h2f, 16, 1, nullptr, 0, wd2, bd2, nullptr, x16, 32, H2, W2, 2, ACT_RELU);
        gru(x16, 32, h3, h3f, 32, g3wz, g3bz, g3wr, g3br, g3wh, g3bh, H3, W3);
        // down 3: 32->64 @ 16x20
        conv(h3f, 32, 1, nullptr, 0, wd3, bd3, nullptr, x16, 64, H3, W3, 2, ACT_RELU);
        gru(x16, 64, h4, h4f, 64, g4wz, g4bz, g4wr, g4br, g4wh, g4bh, H4, W4);

        // decoder: u3 = relu(conv(cat(up2(h4), h3)))  96->32 @ 32x40
        conv(h4f, 64, 2, h3f, 32, wu3, bu3, nullptr, u3f, 32, H3, W3, 1, ACT_RELU);
        // u2 = relu(conv(cat(up2(u3), h2)))  48->16 @ 64x80
        conv(u3f, 32, 2, h2f, 16, wu2, bu2, nullptr, u2f, 16, H2, W2, 1, ACT_RELU);
        // u1 = relu(conv(cat(up2(u2), h1)))  24->8 @ 128x160
        conv(u2f, 16, 2, h1f, 8, wu1, bu1, nullptr, u1f, 8, H_, W_, 1, ACT_RELU);
        // reg = conv(u1) 8->1
        conv(u1f, 8, 1, nullptr, 0, wow, bob, regb, nullptr, 1, H_, W_, 1, ACT_NONE);

        accum_kernel<<<(HW1 + 255) / 256, 256, 0, stream>>>(
            regb, depth_d, esum, dimg, mprob, HW1);
    }

    finalize_kernel<<<(HW1 + 255) / 256, 256, 0, stream>>>(
        dimg, mprob, esum, (float*)d_out, HW1);
}
